// SDR_CNN_FedMoE_22033182228980
// MI455X (gfx1250) — matrix operand order repaired and checked by reading, not compiled, not run on hardware
//
#include <hip/hip_runtime.h>
#include <hip/hip_bf16.h>

typedef __attribute__((ext_vector_type(16))) _Float16 v16h;
typedef __attribute__((ext_vector_type(8)))  float    v8f;
typedef __attribute__((ext_vector_type(4)))  int      v4i;

union AF { v16h v; _Float16 h[16]; };
union CF { v8f  v; float     f[8];  };

#define WMMA_F16(a, b, c) \
  __builtin_amdgcn_wmma_f32_16x16x32_f16(false, (a), false, (b), (short)0, (c), false, false)

// gfx1250 async global->LDS path if this toolchain declares the builtins.
#if defined(__has_builtin)
#if __has_builtin(__builtin_amdgcn_global_load_async_to_lds_b128) && \
    __has_builtin(__builtin_amdgcn_s_wait_asynccnt)
#define USE_ASYNC_LDS 1
#endif
#endif

#ifdef USE_ASYNC_LDS
typedef __attribute__((address_space(1))) v4i* gv4i_p;
typedef __attribute__((address_space(3))) v4i* lv4i_p;
#endif

// ---------- fragment loaders (CDNA5 wave32 WMMA layouts) ----------

// A-matrix 16x32 f16, row-major source, ld in elements. Per lane this is two
// contiguous 16B chunks -> compiler merges to 2x b128 loads.
__device__ inline AF load_a(const _Float16* __restrict__ p, int ld, int lane) {
  AF a;
  int row = lane & 15;
  int hi  = (lane & 16) ? 16 : 0;
  const _Float16* r = p + (size_t)row * ld;
#pragma unroll
  for (int i = 0; i < 8; ++i) {
    int k = 2 * i + hi;
    a.h[2 * i]     = r[k];
    a.h[2 * i + 1] = r[k + 1];
  }
  return a;
}

// B-matrix fragment from fragment-major packed storage: lane's 16 halves are
// 32 contiguous, 32B-aligned bytes -> 2x b128 loads.
__device__ inline AF load_b_contig(const _Float16* __restrict__ p) {
  AF b;
#pragma unroll
  for (int i = 0; i < 16; ++i) b.h[i] = p[i];
  return b;
}

__device__ inline float sigm(float x) { return 1.f / (1.f + expf(-x)); }

// ---------- prep kernels ----------

__global__ __launch_bounds__(256) void k_cast(const float* __restrict__ s,
                                              _Float16* __restrict__ d, int n) {
  int i = blockIdx.x * 256 + threadIdx.x;
  if (i < n) d[i] = (_Float16)s[i];
}

// conv1_w (128,80,7) -> [128][576] f16, zero pad K 560..575
__global__ __launch_bounds__(256) void k_pad_conv1(const float* __restrict__ w,
                                                   _Float16* __restrict__ d) {
  int i = blockIdx.x * 256 + threadIdx.x;
  if (i < 128 * 576) {
    int oc = i / 576, r = i % 576;
    d[i] = (r < 560) ? (_Float16)w[oc * 560 + r] : (_Float16)0.f;
  }
}

__global__ __launch_bounds__(256) void k_badd(const float* __restrict__ a,
                                              const float* __restrict__ b,
                                              float* __restrict__ o, int n) {
  int i = blockIdx.x * 256 + threadIdx.x;
  if (i < n) o[i] = a[i] + b[i];
}

// fold conv bias + BN into y = x*A + B
__global__ __launch_bounds__(256) void k_bnfold(
    const float* __restrict__ cb, const float* __restrict__ g,
    const float* __restrict__ b, const float* __restrict__ m,
    const float* __restrict__ v, float* __restrict__ A, float* __restrict__ B,
    int n) {
  int i = blockIdx.x * 256 + threadIdx.x;
  if (i < n) {
    float s = g[i] * rsqrtf(v[i] + 1e-5f);
    A[i] = s;
    B[i] = (cb[i] - m[i]) * s + b[i];
  }
}

// Pack a KxN B-matrix (f32 source) into fragment-major f16:
// dst[(((kt*ntiles)+nt)*32 + lane)*16 + i] = B[kt*32+k0(lane)+i][nt*16+(lane&15)]
// transposed=1: src is [N][K] (wih-style), else src is [K][N].
__global__ __launch_bounds__(256) void k_packB(const float* __restrict__ src0,
                                               _Float16* __restrict__ dst0,
                                               int K, int N, int transposed,
                                               size_t srcStrideE, size_t dstStrideE) {
  const float* src = src0 + (size_t)blockIdx.y * srcStrideE;
  _Float16* dst = dst0 + (size_t)blockIdx.y * dstStrideE;
  int ntiles = N >> 4;
  int total = (K >> 5) * ntiles * 32;
  int idx = blockIdx.x * 256 + threadIdx.x;
  if (idx >= total) return;
  int lane = idx & 31;
  int t = idx >> 5;
  int nt = t % ntiles, kt = t / ntiles;
  int coln = nt * 16 + (lane & 15);
  int kbase = kt * 32 + ((lane & 16) ? 16 : 0);
  _Float16* d = dst + (size_t)idx * 16;
  for (int i = 0; i < 16; ++i) {
    int k = kbase + i;
    float v = transposed ? src[(size_t)coln * K + k] : src[(size_t)k * N + coln];
    d[i] = (_Float16)v;
  }
}

// ---------- conv1: x(512,80,800) -> y1 f16 [512][128][400] ----------
// Block = 16 conv positions (8 pooled). Implicit GEMM with transposed im2col
// PT[pos][K] built once in LDS -> B fragments are contiguous ds_load_b128.

#define K1PAD 576
__global__ __launch_bounds__(128) void k_conv1(
    const float* __restrict__ x, const _Float16* __restrict__ W1h,
    const float* __restrict__ bnA, const float* __restrict__ bnB,
    _Float16* __restrict__ y1) {
  __shared__ __align__(16) _Float16 Raw[80 * 24];
  __shared__ __align__(16) _Float16 PT[16 * K1PAD];
  int b = blockIdx.y;
  int p0 = blockIdx.x * 16;
  int tid = threadIdx.x, lane = tid & 31, w = tid >> 5;
  for (int idx = tid; idx < 80 * 22; idx += 128) {
    int c = idx / 22, j = idx % 22;
    int gp = p0 + j - 3;
    float v = (gp >= 0 && gp < 800) ? x[((size_t)b * 80 + c) * 800 + gp] : 0.f;
    Raw[c * 24 + j] = (_Float16)v;
  }
  __syncthreads();
  for (int idx = tid; idx < 16 * K1PAD; idx += 128) {
    int pos = idx / K1PAD, r = idx - pos * K1PAD;
    _Float16 v = (_Float16)0.f;
    if (r < 560) {
      int ch = r / 7, kk = r - ch * 7;
      v = Raw[ch * 24 + pos + kk];
    }
    PT[pos * K1PAD + r] = v;
  }
  __syncthreads();
  int col = lane & 15, rbase = (lane & 16) ? 8 : 0;
  const _Float16* pbase = PT + col * K1PAD + ((lane & 16) ? 16 : 0);
  for (int mt = w; mt < 8; mt += 4) {
    CF acc;
    for (int i = 0; i < 8; ++i) acc.f[i] = 0.f;
    for (int ks = 0; ks < 18; ++ks) {
      AF a = load_a(W1h + (size_t)(mt * 16) * 576 + ks * 32, 576, lane);
      AF bf = load_b_contig(pbase + ks * 32);
      acc.v = WMMA_F16(a.v, bf.v, acc.v);
    }
    for (int i = 0; i < 8; ++i) {
      int oc = mt * 16 + rbase + i;
      float vv = fmaxf(acc.f[i] * bnA[oc] + bnB[oc], 0.f);
      float mx = fmaxf(vv, __shfl_xor(vv, 1, 32));
      if (!(lane & 1))
        y1[((size_t)b * 128 + oc) * 400 + blockIdx.x * 8 + (col >> 1)] = (_Float16)mx;
    }
  }
}

// ---------- conv2: y1 -> seq0 f16 [512][200][256] (transposed for LSTM) ----------

#define K2PAD 896
__global__ __launch_bounds__(128) void k_conv2(
    const _Float16* __restrict__ y1, const _Float16* __restrict__ W2h,
    const float* __restrict__ bnA, const float* __restrict__ bnB,
    _Float16* __restrict__ seq0) {
  __shared__ __align__(16) _Float16 Raw[128 * 24];
  __shared__ __align__(16) _Float16 PT[16 * K2PAD];
  int b = blockIdx.y;
  int p0 = blockIdx.x * 16;
  int tid = threadIdx.x, lane = tid & 31, w = tid >> 5;
  for (int idx = tid; idx < 128 * 22; idx += 128) {
    int c = idx / 22, j = idx % 22;
    int gp = p0 + j - 3;
    Raw[c * 24 + j] = (gp >= 0 && gp < 400) ? y1[((size_t)b * 128 + c) * 400 + gp]
                                            : (_Float16)0.f;
  }
  __syncthreads();
  for (int idx = tid; idx < 16 * K2PAD; idx += 128) {
    int pos = idx / K2PAD, r = idx - pos * K2PAD;
    int ch = r / 7, kk = r - ch * 7;
    PT[pos * K2PAD + r] = Raw[ch * 24 + pos + kk];
  }
  __syncthreads();
  int col = lane & 15, rbase = (lane & 16) ? 8 : 0;
  const _Float16* pbase = PT + col * K2PAD + ((lane & 16) ? 16 : 0);
  for (int mt = w; mt < 16; mt += 4) {
    CF acc;
    for (int i = 0; i < 8; ++i) acc.f[i] = 0.f;
    for (int ks = 0; ks < 28; ++ks) {
      AF a = load_a(W2h + (size_t)(mt * 16) * 896 + ks * 32, 896, lane);
      AF bf = load_b_contig(pbase + ks * 32);
      acc.v = WMMA_F16(a.v, bf.v, acc.v);
    }
    for (int i = 0; i < 8; ++i) {
      int oc = mt * 16 + rbase + i;
      float vv = fmaxf(acc.f[i] * bnA[oc] + bnB[oc], 0.f);
      float mx = fmaxf(vv, __shfl_xor(vv, 1, 32));
      if (!(lane & 1)) {
        int P = blockIdx.x * 8 + (col >> 1);
        seq0[((size_t)b * 200 + P) * 256 + oc] = (_Float16)mx;
      }
    }
  }
}

// ---------- LSTM layer: persistent per 16-row batch tile ----------
// Weights are fragment-major packed; xg never materialized (x_t@wihT fused).

#define LDH 264
__global__ __launch_bounds__(128) void k_lstm(
    const _Float16* __restrict__ seq_in, _Float16* __restrict__ seq_out,
    const _Float16* __restrict__ wih_pk, const _Float16* __restrict__ whh_pk,
    const float* __restrict__ bias, float* __restrict__ feat, int write_feat) {
  __shared__ __align__(16) _Float16 Xs[16 * LDH];
  __shared__ __align__(16) _Float16 Hs[2][16 * LDH];
  int b0 = blockIdx.x * 16;
  int tid = threadIdx.x, lane = tid & 31, w = tid >> 5;
  int col = lane & 15, rbase = (lane & 16) ? 8 : 0;
  for (int idx = tid; idx < 16 * LDH; idx += 128) {
    Hs[0][idx] = (_Float16)0.f;
    Hs[1][idx] = (_Float16)0.f;
  }
  float cst[4][8];
  for (int ct = 0; ct < 4; ++ct)
    for (int i = 0; i < 8; ++i) cst[ct][i] = 0.f;
  // per-lane gate biases are time-invariant: load once
  float bsv[4][4];
#pragma unroll
  for (int ct = 0; ct < 4; ++ct)
#pragma unroll
    for (int g = 0; g < 4; ++g)
      bsv[ct][g] = bias[g * 256 + (w * 4 + ct) * 16 + col];

  for (int t = 0; t < 200; ++t) {
    __syncthreads();  // prior step's Hnxt writes + Xs reads complete
    // stage x_t tile: async global->LDS if available, else 16B vector copies
    for (int idx = tid; idx < 16 * 32; idx += 128) {
      int r = idx >> 5, c8 = (idx & 31) * 8;
      const _Float16* gsrc = seq_in + (((size_t)(b0 + r)) * 200 + t) * 256 + c8;
      _Float16* ldst = Xs + r * LDH + c8;
#ifdef USE_ASYNC_LDS
      __builtin_amdgcn_global_load_async_to_lds_b128((gv4i_p)gsrc, (lv4i_p)ldst, 0, 0);
#else
      *(ulonglong2*)ldst = *(const ulonglong2*)gsrc;
#endif
    }
#ifdef USE_ASYNC_LDS
    __builtin_amdgcn_s_wait_asynccnt(0);
#endif
    __syncthreads();
    const _Float16* Hcur = Hs[t & 1];
    _Float16* Hnxt = Hs[(t + 1) & 1];
    for (int ct = 0; ct < 4; ++ct) {
      int wct = w * 4 + ct;           // hidden col-tile 0..15
      int colh = wct * 16;
      __builtin_prefetch(wih_pk + (size_t)wct * 512, 0, 1);
      __builtin_prefetch(whh_pk + (size_t)wct * 512, 0, 1);
      CF accs[4];
#pragma unroll
      for (int g = 0; g < 4; ++g)
        for (int i = 0; i < 8; ++i) accs[g].f[i] = bsv[ct][g];
      for (int ks = 0; ks < 8; ++ks) {
        AF ax = load_a(Xs + ks * 32, LDH, lane);
        AF ah = load_a(Hcur + ks * 32, LDH, lane);
#pragma unroll
        for (int g = 0; g < 4; ++g) {
          // N=1024 -> ntiles=64; nt = g*16 + wct
          size_t fo = (((size_t)ks * 64 + g * 16 + wct) * 32 + lane) * 16;
          AF bx = load_b_contig(wih_pk + fo);
          accs[g].v = WMMA_F16(ax.v, bx.v, accs[g].v);
          AF bh = load_b_contig(whh_pk + fo);
          accs[g].v = WMMA_F16(ah.v, bh.v, accs[g].v);
        }
      }
      for (int i = 0; i < 8; ++i) {
        float ii = sigm(accs[0].f[i]);
        float ff = sigm(accs[1].f[i]);
        float gg = tanhf(accs[2].f[i]);
        float oo = sigm(accs[3].f[i]);
        float c = ff * cst[ct][i] + ii * gg;
        cst[ct][i] = c;
        float h = oo * tanhf(c);
        int r = rbase + i;
        Hnxt[r * LDH + colh + col] = (_Float16)h;
        seq_out[(((size_t)(b0 + r)) * 200 + t) * 256 + colh + col] = (_Float16)h;
        if (write_feat && t == 199)
          feat[(size_t)(b0 + r) * 256 + colh + col] = h;
      }
    }
  }
}

// ---------- MoE + heads ----------

#define LDE 520
__global__ __launch_bounds__(128) void k_moe(
    const float* __restrict__ feat, const float* __restrict__ gate_w,
    const float* __restrict__ gate_b, const float* __restrict__ gbn_g,
    const float* __restrict__ gbn_b, const float* __restrict__ gbn_m,
    const float* __restrict__ gbn_v, const _Float16* __restrict__ ew1,
    const float* __restrict__ eb1, const _Float16* __restrict__ ew2,
    const float* __restrict__ eb2, const float* __restrict__ f1sw,
    const float* __restrict__ f1sb, const float* __restrict__ f2sw,
    const float* __restrict__ f2sb, const float* __restrict__ f3sw,
    const float* __restrict__ f3sb, const float* __restrict__ f1pw,
    const float* __restrict__ f1pb, const float* __restrict__ f2pw,
    const float* __restrict__ f2pb, const float* __restrict__ f3pw,
    const float* __restrict__ f3pb, const float* __restrict__ hw,
    const float* __restrict__ hb, float* __restrict__ out) {
  __shared__ float featF[16 * 256];
  __shared__ __align__(16) _Float16 featH[16 * LDH];
  __shared__ float moe[16 * 256];
  __shared__ __align__(16) _Float16 EHs[16 * LDE];
  __shared__ float logits[16 * 12], gatingS[16 * 12], xsv[16], xpv[16];
  int b0 = blockIdx.x * 16;
  int tid = threadIdx.x, lane = tid & 31, w = tid >> 5;
  int col = lane & 15, rbase = (lane & 16) ? 8 : 0;

  for (int idx = tid; idx < 16 * 256; idx += 128) {
    int r = idx >> 8, ch = idx & 255;
    float v = feat[(size_t)(b0 + r) * 256 + ch];
    featF[idx] = v;
    featH[r * LDH + ch] = (_Float16)v;
    moe[idx] = 0.f;
  }
  __syncthreads();

  // gating: logits + BN + softmax (tiny, VALU)
  for (int idx = tid; idx < 16 * 12; idx += 128) {
    int r = idx / 12, e = idx % 12;
    float s = gate_b[e];
    for (int k = 0; k < 256; ++k) s += featF[r * 256 + k] * gate_w[e * 256 + k];
    s = (s - gbn_m[e]) * rsqrtf(gbn_v[e] + 1e-5f) * gbn_g[e] + gbn_b[e];
    logits[idx] = s;
  }
  __syncthreads();
  if (tid < 16) {
    float mx = -1e30f;
    for (int e = 0; e < 12; ++e) mx = fmaxf(mx, logits[tid * 12 + e]);
    float s = 0.f, ex[12];
    for (int e = 0; e < 12; ++e) { ex[e] = expf(logits[tid * 12 + e] - mx); s += ex[e]; }
    for (int e = 0; e < 12; ++e) gatingS[tid * 12 + e] = ex[e] / s;
  }
  __syncthreads();

  // experts, one at a time per workgroup (WMMA both GEMMs, packed B)
  for (int e = 0; e < 12; ++e) {
    const _Float16* w1 = ew1 + (size_t)e * 131072;  // K=256,N=512 -> ntiles=32
    const _Float16* w2 = ew2 + (size_t)e * 131072;  // K=512,N=256 -> ntiles=16
    for (int nt = w * 8; nt < w * 8 + 8; ++nt) {    // eh = gelu(feat@W1+b1): 16x512
      CF acc;
      for (int i = 0; i < 8; ++i) acc.f[i] = 0.f;
      for (int ks = 0; ks < 8; ++ks) {
        AF a = load_a(featH + ks * 32, LDH, lane);
        AF bf = load_b_contig(w1 + (((size_t)ks * 32 + nt) * 32 + lane) * 16);
        acc.v = WMMA_F16(a.v, bf.v, acc.v);
      }
      for (int i = 0; i < 8; ++i) {
        int cg = nt * 16 + col;
        float xv = acc.f[i] + eb1[e * 512 + cg];
        xv = 0.5f * xv * (1.f + erff(xv * 0.70710678f));
        EHs[(rbase + i) * LDE + cg] = (_Float16)xv;
      }
    }
    __syncthreads();
    for (int nt = w * 4; nt < w * 4 + 4; ++nt) {    // eo = eh@W2+b2, gate-scaled
      CF acc;
      for (int i = 0; i < 8; ++i) acc.f[i] = 0.f;
      for (int ks = 0; ks < 16; ++ks) {
        AF a = load_a(EHs + ks * 32, LDE, lane);
        AF bf = load_b_contig(w2 + (((size_t)ks * 16 + nt) * 32 + lane) * 16);
        acc.v = WMMA_F16(a.v, bf.v, acc.v);
      }
      for (int i = 0; i < 8; ++i) {
        int r = rbase + i, co = nt * 16 + col;
        moe[r * 256 + co] += (acc.f[i] + eb2[e * 256 + co]) * gatingS[r * 12 + e];
      }
    }
    __syncthreads();
  }

  // heads (tiny FC stacks) — reuse EHs storage as float scratch
  float* H1 = (float*)EHs;            // 16*128
  float* H2 = H1 + 16 * 128;          // 16*64
  for (int idx = tid; idx < 16 * 128; idx += 128) {
    int r = idx >> 7, o = idx & 127;
    float s = f1sb[o];
    for (int k = 0; k < 256; ++k) s += moe[r * 256 + k] * f1sw[o * 256 + k];
    H1[idx] = fmaxf(s, 0.f);
  }
  __syncthreads();
  for (int idx = tid; idx < 16 * 64; idx += 128) {
    int r = idx >> 6, o = idx & 63;
    float s = f2sb[o];
    for (int k = 0; k < 128; ++k) s += H1[r * 128 + k] * f2sw[o * 128 + k];
    H2[idx] = fmaxf(s, 0.f);
  }
  __syncthreads();
  if (tid < 16) {
    float s = f3sb[0];
    for (int k = 0; k < 64; ++k) s += H2[tid * 64 + k] * f3sw[k];
    xsv[tid] = s;
  }
  __syncthreads();
  for (int idx = tid; idx < 16 * 128; idx += 128) {
    int r = idx >> 7, o = idx & 127;
    float s = f1pb[o];
    for (int k = 0; k < 256; ++k) s += moe[r * 256 + k] * f1pw[o * 256 + k];
    H1[idx] = fmaxf(s, 0.f);
  }
  __syncthreads();
  for (int idx = tid; idx < 16 * 64; idx += 128) {
    int r = idx >> 6, o = idx & 63;
    float s = f2pb[o];
    for (int k = 0; k < 128; ++k) s += H1[r * 128 + k] * f2pw[o * 128 + k];
    H2[idx] = fmaxf(s, 0.f);
  }
  __syncthreads();
  if (tid < 16) {
    float s = f3pb[0];
    for (int k = 0; k < 64; ++k) s += H2[tid * 64 + k] * f3pw[k];
    xpv[tid] = s;
    float g0 = hb[0], g1 = hb[1];
    for (int k = 0; k < 256; ++k) {
      g0 += moe[tid * 256 + k] * hw[k];
      g1 += moe[tid * 256 + k] * hw[256 + k];
    }
    float mx = fmaxf(g0, g1);
    float e0 = expf(g0 - mx), e1 = expf(g1 - mx);
    float inv = 1.f / (e0 + e1);
    float val = (e0 * inv) * xsv[tid] + (e1 * inv) * xpv[tid];
    out[b0 + tid] = 1.f / (1.f + expf(-val));
  }
}

// ---------- launcher ----------

extern "C" void kernel_launch(void* const* d_in, const int* in_sizes, int n_in,
                              void* d_out, int out_size, void* d_ws, size_t ws_size,
                              hipStream_t stream) {
  (void)in_sizes; (void)n_in; (void)out_size; (void)ws_size;
  const float* x       = (const float*)d_in[0];
  const float* conv1_w = (const float*)d_in[1];
  const float* conv1_b = (const float*)d_in[2];
  const float* bn1_g   = (const float*)d_in[3];
  const float* bn1_b   = (const float*)d_in[4];
  const float* bn1_m   = (const float*)d_in[5];
  const float* bn1_v   = (const float*)d_in[6];
  const float* conv2_w = (const float*)d_in[7];
  const float* conv2_b = (const float*)d_in[8];
  const float* bn2_g   = (const float*)d_in[9];
  const float* bn2_b   = (const float*)d_in[10];
  const float* bn2_m   = (const float*)d_in[11];
  const float* bn2_v   = (const float*)d_in[12];
  const float* wih0    = (const float*)d_in[13];
  const float* whh0    = (const float*)d_in[14];
  const float* bih0    = (const float*)d_in[15];
  const float* bhh0    = (const float*)d_in[16];
  const float* wih1    = (const float*)d_in[17];
  const float* whh1    = (const float*)d_in[18];
  const float* bih1    = (const float*)d_in[19];
  const float* bhh1    = (const float*)d_in[20];
  const float* gate_w  = (const float*)d_in[21];
  const float* gate_b  = (const float*)d_in[22];
  const float* gbn_g   = (const float*)d_in[23];
  const float* gbn_b   = (const float*)d_in[24];
  const float* gbn_m   = (const float*)d_in[25];
  const float* gbn_v   = (const float*)d_in[26];
  const float* exp_w1  = (const float*)d_in[27];
  const float* exp_b1  = (const float*)d_in[28];
  const float* exp_w2  = (const float*)d_in[29];
  const float* exp_b2  = (const float*)d_in[30];
  const float* f1sw = (const float*)d_in[31]; const float* f1sb = (const float*)d_in[32];
  const float* f2sw = (const float*)d_in[33]; const float* f2sb = (const float*)d_in[34];
  const float* f3sw = (const float*)d_in[35]; const float* f3sb = (const float*)d_in[36];
  const float* f1pw = (const float*)d_in[37]; const float* f1pb = (const float*)d_in[38];
  const float* f2pw = (const float*)d_in[39]; const float* f2pb = (const float*)d_in[40];
  const float* f3pw = (const float*)d_in[41]; const float* f3pb = (const float*)d_in[42];
  const float* head_w = (const float*)d_in[43];
  const float* head_b = (const float*)d_in[44];

  char* ws = (char*)d_ws;
  size_t off = 0;
  auto alloc = [&](size_t bytes) {
    void* p = ws + off;
    off += (bytes + 255) & ~(size_t)255;
    return p;
  };
  _Float16* W1h    = (_Float16*)alloc((size_t)128 * 576 * 2);
  _Float16* W2h    = (_Float16*)alloc((size_t)256 * 896 * 2);
  _Float16* wih0pk = (_Float16*)alloc((size_t)256 * 1024 * 2);
  _Float16* whh0pk = (_Float16*)alloc((size_t)256 * 1024 * 2);
  _Float16* wih1pk = (_Float16*)alloc((size_t)256 * 1024 * 2);
  _Float16* whh1pk = (_Float16*)alloc((size_t)256 * 1024 * 2);
  float*    bias0  = (float*)alloc(1024 * 4);
  float*    bias1  = (float*)alloc(1024 * 4);
  float*    bnA1   = (float*)alloc(128 * 4);
  float*    bnB1   = (float*)alloc(128 * 4);
  float*    bnA2   = (float*)alloc(256 * 4);
  float*    bnB2   = (float*)alloc(256 * 4);
  _Float16* ew1pk  = (_Float16*)alloc((size_t)12 * 256 * 512 * 2);
  _Float16* ew2pk  = (_Float16*)alloc((size_t)12 * 512 * 256 * 2);
  float*    featB  = (float*)alloc((size_t)512 * 256 * 4);
  _Float16* bufA   = (_Float16*)alloc((size_t)512 * 128 * 400 * 2);  // y1, then seq1
  _Float16* bufB   = (_Float16*)alloc((size_t)512 * 200 * 256 * 2);  // seq0, then dummy

  auto blk = [](int n) { return (n + 255) / 256; };
  // weight prep
  k_pad_conv1<<<blk(128 * 576), 256, 0, stream>>>(conv1_w, W1h);
  k_cast<<<blk(256 * 896), 256, 0, stream>>>(conv2_w, W2h, 256 * 896);
  k_packB<<<blk(16384), 256, 0, stream>>>(wih0, wih0pk, 256, 1024, 1, 0, 0);
  k_packB<<<blk(16384), 256, 0, stream>>>(whh0, whh0pk, 256, 1024, 1, 0, 0);
  k_packB<<<blk(16384), 256, 0, stream>>>(wih1, wih1pk, 256, 1024, 1, 0, 0);
  k_packB<<<blk(16384), 256, 0, stream>>>(whh1, whh1pk, 256, 1024, 1, 0, 0);
  k_packB<<<dim3(blk(8192), 12), 256, 0, stream>>>(exp_w1, ew1pk, 256, 512, 0,
                                                   (size_t)256 * 512, (size_t)131072);
  k_packB<<<dim3(blk(8192), 12), 256, 0, stream>>>(exp_w2, ew2pk, 512, 256, 0,
                                                   (size_t)512 * 256, (size_t)131072);
  k_badd<<<blk(1024), 256, 0, stream>>>(bih0, bhh0, bias0, 1024);
  k_badd<<<blk(1024), 256, 0, stream>>>(bih1, bhh1, bias1, 1024);
  k_bnfold<<<1, 256, 0, stream>>>(conv1_b, bn1_g, bn1_b, bn1_m, bn1_v, bnA1, bnB1, 128);
  k_bnfold<<<1, 256, 0, stream>>>(conv2_b, bn2_g, bn2_b, bn2_m, bn2_v, bnA2, bnB2, 256);

  // pipeline
  k_conv1<<<dim3(50, 512), 128, 0, stream>>>(x, W1h, bnA1, bnB1, bufA);
  k_conv2<<<dim3(25, 512), 128, 0, stream>>>(bufA, W2h, bnA2, bnB2, bufB);
  k_lstm<<<32, 128, 0, stream>>>(bufB, bufA, wih0pk, whh0pk, bias0, featB, 0);
  k_lstm<<<32, 128, 0, stream>>>(bufA, bufB, wih1pk, whh1pk, bias1, featB, 1);
  k_moe<<<32, 128, 0, stream>>>(featB, gate_w, gate_b, gbn_g, gbn_b, gbn_m, gbn_v,
                                ew1pk, exp_b1, ew2pk, exp_b2,
                                f1sw, f1sb, f2sw, f2sb, f3sw, f3sb,
                                f1pw, f1pb, f2pw, f2pb, f3pw, f3pb,
                                head_w, head_b, (float*)d_out);
}
